// ChannelTransformerReceiver_56753697849661
// MI455X (gfx1250) — matrix operand order repaired and checked by reading, not compiled, not run on hardware
//
#include <hip/hip_runtime.h>

// ---------------------------------------------------------------------------
// CDNA5 (gfx1250, wave32) implementation of ChannelTransformerReceiver.
// All dense math runs through v_wmma_f32_16x16x32_bf16 (bf16 in, f32 acc).
// LDS tiles are laid out so every WMMA fragment load is contiguous 128-bit.
// ---------------------------------------------------------------------------

typedef __bf16 bf16_t;
typedef __attribute__((ext_vector_type(16))) __bf16 v16bf;
typedef __attribute__((ext_vector_type(8)))  __bf16 v8bf;
typedef __attribute__((ext_vector_type(8)))  float  v8f;

#define EPI_GELU       1
#define EPI_STORE_F32  2
#define EPI_STORE_BF16 4

// problem dims
#define CB    256   // batch
#define CDF   512   // d_in
#define CD    512   // d_model
#define CN    128   // carriers (seq len)
#define CL    4     // layers
#define CH    8     // heads
#define CHD   64    // head dim
#define CDFF  2048
#define CHID  272
#define COUT  32
#define MTOK  (CB * CN)   // 32768 tokens

static __device__ __forceinline__ bf16_t f2bf(float f) {
  unsigned int u = __builtin_bit_cast(unsigned int, f);
  unsigned int lsb = (u >> 16) & 1u;          // round-to-nearest-even
  u += 0x7FFFu + lsb;
  return __builtin_bit_cast(bf16_t, (unsigned short)(u >> 16));
}
static __device__ __forceinline__ float gelu_exact(float x) {
  return 0.5f * x * (1.0f + erff(x * 0.70710678118654752f));
}
static __device__ __forceinline__ v8f zero_v8f() {
  v8f z = {0.f, 0.f, 0.f, 0.f, 0.f, 0.f, 0.f, 0.f};
  return z;
}

// --- WMMA fragment loaders (contiguous k = 16*half + i order, same for A and B; wave32) ----------
// A fragment: 16x32 bf16 from row-major tile (stride in elements, 16B-mult).
// lanes 0-15 : M=lane,    K groups {0..7, 8..15}
// lanes 16-31: M=lane-16, K groups {16..23, 24..31}
// -> two contiguous 8-element (16B) reads per lane.
static __device__ __forceinline__ v16bf frag_a_ld(const bf16_t* base, int stride, int lane) {
  int m = lane & 15;
  int half = lane >> 4;
  int g0 = half ? 16 : 0;
  int g1 = half ? 24 : 8;
  const bf16_t* row = base + m * stride;
  v16bf f;
#pragma unroll
  for (int i = 0; i < 8; ++i) f[i] = row[g0 + i];
#pragma unroll
  for (int i = 0; i < 8; ++i) f[8 + i] = row[g1 + i];
  return f;
}
// B fragment (32x16): logical B(k,n) stored n-major: tile[n][k].
// lanes 0-15: N=lane, K=0..15 ; lanes 16-31: N=lane-16, K=16..31
// -> one contiguous 16-element (32B) read per lane.
static __device__ __forceinline__ v16bf frag_bt_ld(const bf16_t* base, int stride, int lane) {
  int n = lane & 15;
  int kb = (lane >> 4) << 4;
  const bf16_t* p = base + n * stride + kb;
  v16bf f;
#pragma unroll
  for (int i = 0; i < 16; ++i) f[i] = p[i];
  return f;
}
#define WMMA_BF16(a, b, c) \
  __builtin_amdgcn_wmma_f32_16x16x32_bf16(false, (a), false, (b), (short)0, (c), false, false)

// ---------------------------------------------------------------------------
// Generic tiled WMMA GEMM: C[z] = A[z](MxK) * W[z](KxN) + bias[z]
// Block tile 128x64, K step 32, 8 waves (4x2), each wave a 32x32 sub-tile.
// EDGE=false: M%128==0, K%32==0, N%64==0 -> unguarded b128 staging loads.
// ---------------------------------------------------------------------------
#define GM_BM 128
#define GM_BN 64
#define GM_BK 32
#define GM_AST (GM_BK + 8)   // 40 elems = 80B (16B multiple)
#define GM_BST (GM_BK + 8)   // Bst is [BN][BK+8], transposed

template <bool EDGE>
__global__ __launch_bounds__(256) void wmma_gemm_bf16_kernel(
    const bf16_t* __restrict__ A, long long lda, long long aBatch,
    const bf16_t* __restrict__ W, long long wBatch,
    const float*  __restrict__ bias, long long biasBatch,
    float* __restrict__ Cf, bf16_t* __restrict__ Cb,
    long long ldc, long long cBatch,
    int M, int K, int Nout, int flags)
{
  __shared__ bf16_t As[GM_BM][GM_AST];
  __shared__ bf16_t Bst[GM_BN][GM_BST];   // Bst[n][k] = W[k0+k][nblk+n]

  const int tid  = threadIdx.x;
  const int lane = tid & 31;
  const int w    = tid >> 5;
  const int wm   = w >> 1;         // 0..3 -> 32-row strip
  const int wn   = w & 1;          // 0..1 -> 32-col strip
  const int mblk = blockIdx.x * GM_BM;
  const int nblk = blockIdx.y * GM_BN;
  const int z    = blockIdx.z;

  const bf16_t* Az = A + (long long)z * aBatch;
  const bf16_t* Wz = W + (long long)z * wBatch;
  const float*  bz = bias ? bias + (long long)z * biasBatch : nullptr;
  float*  cfz = Cf ? Cf + (long long)z * cBatch : nullptr;
  bf16_t* cbz = Cb ? Cb + (long long)z * cBatch : nullptr;

  const bf16_t bzero = __builtin_bit_cast(bf16_t, (unsigned short)0);
  v8f acc[2][2] = {{zero_v8f(), zero_v8f()}, {zero_v8f(), zero_v8f()}};

  for (int k0 = 0; k0 < K; k0 += GM_BK) {
    if (!EDGE) {
      // A tile: 512 x 16B chunks, 2 per thread, fully coalesced.
#pragma unroll
      for (int it = 0; it < 2; ++it) {
        int chunk = tid + it * 256;
        int r = chunk >> 2, c8 = (chunk & 3) * 8;
        *(v8bf*)&As[r][c8] =
            *(const v8bf*)(Az + (size_t)(mblk + r) * lda + (k0 + c8));
      }
      // B tile: 256 x 16B chunks, 1 per thread; scatter-transpose into LDS.
      {
        int kq = tid >> 3, c8 = (tid & 7) * 8;
        v8bf wv = *(const v8bf*)(Wz + (size_t)(k0 + kq) * Nout + (nblk + c8));
#pragma unroll
        for (int i = 0; i < 8; ++i) Bst[c8 + i][kq] = wv[i];
      }
    } else {
      // guarded path: clamped-address load + select (no exec-mask branches)
      for (int e = tid; e < GM_BM * GM_BK; e += 256) {
        int r = e >> 5, c = e & 31;
        int gm = mblk + r, gk = k0 + c;
        bool ok = (gm < M) & (gk < K);
        size_t ai = (size_t)(ok ? gm : 0) * lda + (ok ? gk : 0);
        bf16_t v = Az[ai];
        As[r][c] = ok ? v : bzero;
      }
      for (int e = tid; e < GM_BK * GM_BN; e += 256) {
        int kq = e >> 6, c = e & 63;
        int gk = k0 + kq, gn = nblk + c;
        bool ok = (gk < K) & (gn < Nout);
        size_t wi = (size_t)(ok ? gk : 0) * Nout + (ok ? gn : 0);
        bf16_t v = Wz[wi];
        Bst[c][kq] = ok ? v : bzero;
      }
    }
    __syncthreads();

    v16bf a0 = frag_a_ld(&As[wm * 32][0],       GM_AST, lane);
    v16bf a1 = frag_a_ld(&As[wm * 32 + 16][0],  GM_AST, lane);
    v16bf b0 = frag_bt_ld(&Bst[wn * 32][0],      GM_BST, lane);
    v16bf b1 = frag_bt_ld(&Bst[wn * 32 + 16][0], GM_BST, lane);
    acc[0][0] = WMMA_BF16(a0, b0, acc[0][0]);
    acc[0][1] = WMMA_BF16(a0, b1, acc[0][1]);
    acc[1][0] = WMMA_BF16(a1, b0, acc[1][0]);
    acc[1][1] = WMMA_BF16(a1, b1, acc[1][1]);
    __syncthreads();
  }

  // epilogue: C/D 16x16 layout: VGPR r -> M=r+8*half, N=lane&15
  const int nn   = lane & 15;
  const int half = lane >> 4;
#pragma unroll
  for (int i = 0; i < 2; ++i) {
#pragma unroll
    for (int j = 0; j < 2; ++j) {
      int gn = nblk + wn * 32 + j * 16 + nn;
      if (EDGE && gn >= Nout) continue;
      float bv = bz ? bz[gn] : 0.f;
      int gm0 = mblk + wm * 32 + i * 16 + half * 8;
#pragma unroll
      for (int r = 0; r < 8; ++r) {
        int gm = gm0 + r;
        if (EDGE && gm >= M) continue;
        float v = acc[i][j][r] + bv;
        if (flags & EPI_GELU) v = gelu_exact(v);
        if (flags & EPI_STORE_F32)  cfz[(size_t)gm * ldc + gn] = v;
        if (flags & EPI_STORE_BF16) cbz[(size_t)gm * ldc + gn] = f2bf(v);
      }
    }
  }
}

// ---------------------------------------------------------------------------
// Fused attention: one block per (b,h). N=128 keys, hd=64.
// Each wave owns 16 full query rows -> softmax reduces with shuffles only.
// V is staged transposed so the P@V B-fragments are contiguous b128 loads.
// ---------------------------------------------------------------------------
#define AT_ST  (CHD + 8)   // Q/K LDS row stride: 72 elems = 144B (16B mult)
#define VST_ST (CN + 8)    // Vst/P LDS row stride: 136 elems = 272B (16B mult)
#define PS_ST  (CN + 8)

__global__ __launch_bounds__(256) void attention_kernel(
    const bf16_t* __restrict__ Qg, const bf16_t* __restrict__ Kg,
    const bf16_t* __restrict__ Vg, bf16_t* __restrict__ Og)
{
  __shared__ bf16_t Qs[CN][AT_ST];
  __shared__ bf16_t Ks[CN][AT_ST];
  __shared__ bf16_t Vst[CHD][VST_ST];   // Vst[hd][key]
  __shared__ bf16_t Ps[CN][PS_ST];

  const int b = blockIdx.x >> 3;
  const int h = blockIdx.x & 7;
  const size_t base = (size_t)b * CN * CD + (size_t)h * CHD;

  const int tid  = threadIdx.x;
  const int lane = tid & 31;
  const int w    = tid >> 5;          // wave owns query rows [w*16, w*16+16)
  const int nn   = lane & 15;
  const int half = lane >> 4;

  // stage Q,K (row-major) and V (transposed): 1024 x 16B chunks each
#pragma unroll
  for (int it = 0; it < 4; ++it) {
    int chunk = tid + it * 256;
    int r = chunk >> 3, c8 = (chunk & 7) * 8;
    size_t g = base + (size_t)r * CD + c8;
    *(v8bf*)&Qs[r][c8] = *(const v8bf*)(Qg + g);
    *(v8bf*)&Ks[r][c8] = *(const v8bf*)(Kg + g);
    v8bf vv = *(const v8bf*)(Vg + g);
#pragma unroll
    for (int i = 0; i < 8; ++i) Vst[c8 + i][r] = vv[i];
  }
  __syncthreads();

  // S = Q @ K^T : 8 col-tiles of 16 keys, K-dim 64 in 2 steps
  v8f sc[8];
#pragma unroll
  for (int t = 0; t < 8; ++t) sc[t] = zero_v8f();
#pragma unroll
  for (int ks = 0; ks < CHD; ks += 32) {
    v16bf aq = frag_a_ld(&Qs[w * 16][ks], AT_ST, lane);
#pragma unroll
    for (int t = 0; t < 8; ++t) {
      v16bf bk = frag_bt_ld(&Ks[t * 16][ks], AT_ST, lane);
      sc[t] = WMMA_BF16(aq, bk, sc[t]);
    }
  }

  // softmax over 128 columns (row lives in 16 lanes of one half + 8 tiles)
  float mx[8], sm[8];
#pragma unroll
  for (int r = 0; r < 8; ++r) mx[r] = -1e30f;
#pragma unroll
  for (int t = 0; t < 8; ++t)
#pragma unroll
    for (int r = 0; r < 8; ++r) {
      float v = sc[t][r] * 0.125f;        // 1/sqrt(64)
      sc[t][r] = v;
      mx[r] = fmaxf(mx[r], v);
    }
#pragma unroll
  for (int off = 1; off < 16; off <<= 1)
#pragma unroll
    for (int r = 0; r < 8; ++r) mx[r] = fmaxf(mx[r], __shfl_xor(mx[r], off, 32));
#pragma unroll
  for (int r = 0; r < 8; ++r) sm[r] = 0.f;
#pragma unroll
  for (int t = 0; t < 8; ++t)
#pragma unroll
    for (int r = 0; r < 8; ++r) {
      float e = __expf(sc[t][r] - mx[r]);
      sc[t][r] = e;
      sm[r] += e;
    }
#pragma unroll
  for (int off = 1; off < 16; off <<= 1)
#pragma unroll
    for (int r = 0; r < 8; ++r) sm[r] += __shfl_xor(sm[r], off, 32);
#pragma unroll
  for (int r = 0; r < 8; ++r) sm[r] = 1.0f / sm[r];
#pragma unroll
  for (int t = 0; t < 8; ++t)
#pragma unroll
    for (int r = 0; r < 8; ++r)
      Ps[w * 16 + half * 8 + r][t * 16 + nn] = f2bf(sc[t][r] * sm[r]);
  __syncthreads();

  // O = P @ V : 4 col-tiles of 16, K-dim 128 in 4 steps
  v8f oc[4];
#pragma unroll
  for (int j = 0; j < 4; ++j) oc[j] = zero_v8f();
#pragma unroll
  for (int ks = 0; ks < CN; ks += 32) {
    v16bf ap = frag_a_ld(&Ps[w * 16][ks], PS_ST, lane);
#pragma unroll
    for (int j = 0; j < 4; ++j) {
      v16bf bv = frag_bt_ld(&Vst[j * 16][ks], VST_ST, lane);
      oc[j] = WMMA_BF16(ap, bv, oc[j]);
    }
  }
#pragma unroll
  for (int j = 0; j < 4; ++j)
#pragma unroll
    for (int r = 0; r < 8; ++r) {
      int row = w * 16 + half * 8 + r;
      Og[base + (size_t)row * CD + j * 16 + nn] = f2bf(oc[j][r]);
    }
}

// ---------------------------------------------------------------------------
// Residual add + LayerNorm (biased var), one wave per 512-wide row.
// ---------------------------------------------------------------------------
__global__ __launch_bounds__(256) void ln_residual_kernel(
    float* __restrict__ x, const float* __restrict__ res,
    const float* __restrict__ g, const float* __restrict__ beta,
    bf16_t* __restrict__ xb, int rows)
{
  int row  = blockIdx.x * 8 + (threadIdx.x >> 5);
  int lane = threadIdx.x & 31;
  if (row >= rows) return;
  float* xr = x + (size_t)row * CD;
  const float* rr = res + (size_t)row * CD;

  float vals[16];
  float s = 0.f;
#pragma unroll
  for (int i = 0; i < 16; ++i) {
    int c = lane + i * 32;
    vals[i] = xr[c] + rr[c];
    s += vals[i];
  }
#pragma unroll
  for (int off = 16; off; off >>= 1) s += __shfl_xor(s, off, 32);
  float mean = s * (1.0f / CD);
  float vs = 0.f;
#pragma unroll
  for (int i = 0; i < 16; ++i) {
    float d = vals[i] - mean;
    vs += d * d;
  }
#pragma unroll
  for (int off = 16; off; off >>= 1) vs += __shfl_xor(vs, off, 32);
  float rstd = rsqrtf(vs * (1.0f / CD) + 1e-5f);
#pragma unroll
  for (int i = 0; i < 16; ++i) {
    int c = lane + i * 32;
    float y = (vals[i] - mean) * rstd * g[c] + beta[c];
    xr[c] = y;
    xb[(size_t)row * CD + c] = f2bf(y);
  }
}

// ---------------------------------------------------------------------------
// Elementwise helpers
// ---------------------------------------------------------------------------
__global__ void f32_to_bf16_kernel(const float* __restrict__ in,
                                   bf16_t* __restrict__ out, size_t n) {
  size_t i = (size_t)blockIdx.x * 256 + threadIdx.x;
  if (i < n) out[i] = f2bf(in[i]);
}

// x[b,n,:] = adapt[b,:] + pos[n,:]
__global__ void broadcast_pos_kernel(const float* __restrict__ adapt,
                                     const float* __restrict__ pos,
                                     float* __restrict__ xf,
                                     bf16_t* __restrict__ xb) {
  size_t idx = (size_t)blockIdx.x * 256 + threadIdx.x;
  if (idx >= (size_t)MTOK * CD) return;
  int d = (int)(idx & (CD - 1));
  int n = (int)((idx >> 9) & (CN - 1));
  int b = (int)(idx >> 16);
  float v = adapt[(size_t)b * CD + d] + pos[(size_t)n * CD + d];
  xf[idx] = v;
  xb[idx] = f2bf(v);
}

// BatchNorm training-mode stats over B for each (n,c)
__global__ void bn_stats_kernel(const float* __restrict__ h,
                                float* __restrict__ mean,
                                float* __restrict__ rstd, int NC, int Brows) {
  int i = blockIdx.x * 256 + threadIdx.x;
  if (i >= NC) return;
  float s = 0.f, s2 = 0.f;
  for (int b = 0; b < Brows; ++b) {
    float v = h[(size_t)b * NC + i];
    s += v;
    s2 += v * v;
  }
  float m = s / Brows;
  float var = s2 / Brows - m * m;
  mean[i] = m;
  rstd[i] = rsqrtf(var + 1e-5f);
}

__global__ void bn_apply_gelu_kernel(const float* __restrict__ h,
                                     const float* __restrict__ mean,
                                     const float* __restrict__ rstd,
                                     const float* __restrict__ g,
                                     const float* __restrict__ bb,
                                     bf16_t* __restrict__ out,
                                     int NC, size_t total) {
  size_t idx = (size_t)blockIdx.x * 256 + threadIdx.x;
  if (idx >= total) return;
  int i = (int)(idx % (size_t)NC);
  float v = (h[idx] - mean[i]) * rstd[i] * g[i] + bb[i];
  out[idx] = f2bf(gelu_exact(v));
}

// [B,N,32] -> [B, N_RX=4, N_TX=8, N] : out[b,rx,tx,n] = o[(b*N+n)*32 + tx*4+rx]
__global__ void out_transpose_kernel(const float* __restrict__ o,
                                     float* __restrict__ out) {
  size_t idx = (size_t)blockIdx.x * 256 + threadIdx.x;
  if (idx >= (size_t)CB * COUT * CN) return;
  int n  = (int)(idx & 127);
  int tx = (int)((idx >> 7) & 7);
  int rx = (int)((idx >> 10) & 3);
  int b  = (int)(idx >> 12);
  out[idx] = o[((size_t)b * CN + n) * COUT + tx * 4 + rx];
}

// ---------------------------------------------------------------------------
// Host orchestration
// ---------------------------------------------------------------------------
extern "C" void kernel_launch(void* const* d_in, const int* in_sizes, int n_in,
                              void* d_out, int out_size, void* d_ws, size_t ws_size,
                              hipStream_t stream) {
  const float* input  = (const float*)d_in[0];
  const float* Win    = (const float*)d_in[1];
  const float* bin_   = (const float*)d_in[2];
  const float* pos    = (const float*)d_in[3];
  const float* Wq     = (const float*)d_in[4];
  const float* bq     = (const float*)d_in[5];
  const float* Wk     = (const float*)d_in[6];
  const float* bk     = (const float*)d_in[7];
  const float* Wv     = (const float*)d_in[8];
  const float* bv     = (const float*)d_in[9];
  const float* Wo     = (const float*)d_in[10];
  const float* bo     = (const float*)d_in[11];
  const float* ln1_g  = (const float*)d_in[12];
  const float* ln1_b  = (const float*)d_in[13];
  const float* W1     = (const float*)d_in[14];
  const float* b1     = (const float*)d_in[15];
  const float* W2f    = (const float*)d_in[16];
  const float* b2f    = (const float*)d_in[17];
  const float* ln2_g  = (const float*)d_in[18];
  const float* ln2_b  = (const float*)d_in[19];
  const float* mW1    = (const float*)d_in[20];
  const float* mb1    = (const float*)d_in[21];
  const float* bn1_g  = (const float*)d_in[22];
  const float* bn1_b  = (const float*)d_in[23];
  const float* mW2    = (const float*)d_in[24];
  const float* mb2    = (const float*)d_in[25];
  const float* bn2_g  = (const float*)d_in[26];
  const float* bn2_b  = (const float*)d_in[27];
  const float* mW3    = (const float*)d_in[28];
  const float* mb3    = (const float*)d_in[29];

  char* ws = (char*)d_ws;
  size_t off = 0;
  auto take = [&](size_t bytes) -> char* {
    char* p = ws + off;
    off += (bytes + 255) & ~(size_t)255;
    return p;
  };

  // bf16 weight copies
  bf16_t* in_bf = (bf16_t*)take((size_t)CB * CDF * 2);
  bf16_t* wWin  = (bf16_t*)take((size_t)CDF * CD * 2);
  bf16_t* wQ    = (bf16_t*)take((size_t)CL * CD * CD * 2);
  bf16_t* wK    = (bf16_t*)take((size_t)CL * CD * CD * 2);
  bf16_t* wV    = (bf16_t*)take((size_t)CL * CD * CD * 2);
  bf16_t* wO    = (bf16_t*)take((size_t)CL * CD * CD * 2);
  bf16_t* wW1   = (bf16_t*)take((size_t)CL * CD * CDFF * 2);
  bf16_t* wW2   = (bf16_t*)take((size_t)CL * CDFF * CD * 2);
  bf16_t* wM1   = (bf16_t*)take((size_t)CN * CD * CHID * 2);
  bf16_t* wM2   = (bf16_t*)take((size_t)CN * CHID * CHID * 2);
  bf16_t* wM3   = (bf16_t*)take((size_t)CN * CHID * COUT * 2);
  // activations
  float*  adapt32 = (float*)take((size_t)CB * CD * 4);
  float*  xf      = (float*)take((size_t)MTOK * CD * 4);
  bf16_t* xb      = (bf16_t*)take((size_t)MTOK * CD * 2);
  bf16_t* qb      = (bf16_t*)take((size_t)MTOK * CD * 2);
  bf16_t* kb_     = (bf16_t*)take((size_t)MTOK * CD * 2);
  bf16_t* vb_     = (bf16_t*)take((size_t)MTOK * CD * 2);
  bf16_t* ctxb    = (bf16_t*)take((size_t)MTOK * CD * 2);
  float*  tmp32   = (float*)take((size_t)MTOK * CD * 4);
  char*   big     = take((size_t)MTOK * CDFF * 2);   // ffn hidden, later head bufs
  float*  mean_   = (float*)take((size_t)CN * CHID * 4);
  float*  rstd_   = (float*)take((size_t)CN * CHID * 4);

  // head buffers reuse "big" / tmp32 / qb after the transformer finishes
  bf16_t* ffn_bf = (bf16_t*)big;
  float*  h1f = (float*)big;                                   // 35.7 MB
  bf16_t* h1b = (bf16_t*)(big + (size_t)35651584);             // 17.8 MB
  bf16_t* h2b = (bf16_t*)(big + (size_t)35651584 + 17825792);  // 17.8 MB
  float*  h2f = tmp32;
  float*  hout = (float*)qb;

  dim3 blk(256);
  auto conv = [&](const float* src, bf16_t* dst, size_t n) {
    f32_to_bf16_kernel<<<(unsigned)((n + 255) / 256), blk, 0, stream>>>(src, dst, n);
  };
  auto gemm = [&](const bf16_t* A, long long lda, long long aB,
                  const bf16_t* W, long long wB,
                  const float* bias, long long bB,
                  float* Cf, bf16_t* Cb, long long ldc, long long cB,
                  int M, int K, int Nout, int flags, int zc, bool edge) {
    dim3 g((unsigned)((M + GM_BM - 1) / GM_BM),
           (unsigned)((Nout + GM_BN - 1) / GM_BN), (unsigned)zc);
    if (edge)
      wmma_gemm_bf16_kernel<true><<<g, blk, 0, stream>>>(
          A, lda, aB, W, wB, bias, bB, Cf, Cb, ldc, cB, M, K, Nout, flags);
    else
      wmma_gemm_bf16_kernel<false><<<g, blk, 0, stream>>>(
          A, lda, aB, W, wB, bias, bB, Cf, Cb, ldc, cB, M, K, Nout, flags);
  };

  // ---- convert everything we feed to WMMA into bf16 ----
  conv(input, in_bf, (size_t)CB * CDF);
  conv(Win,  wWin, (size_t)CDF * CD);
  conv(Wq,   wQ,   (size_t)CL * CD * CD);
  conv(Wk,   wK,   (size_t)CL * CD * CD);
  conv(Wv,   wV,   (size_t)CL * CD * CD);
  conv(Wo,   wO,   (size_t)CL * CD * CD);
  conv(W1,   wW1,  (size_t)CL * CD * CDFF);
  conv(W2f,  wW2,  (size_t)CL * CDFF * CD);
  conv(mW1,  wM1,  (size_t)CN * CD * CHID);
  conv(mW2,  wM2,  (size_t)CN * CHID * CHID);
  conv(mW3,  wM3,  (size_t)CN * CHID * COUT);

  // ---- input adapter + positional broadcast ----
  gemm(in_bf, CDF, 0, wWin, 0, bin_, 0, adapt32, nullptr, CD, 0,
       CB, CDF, CD, EPI_STORE_F32, 1, false);
  broadcast_pos_kernel<<<(unsigned)(((size_t)MTOK * CD) / 256), blk, 0, stream>>>(
      adapt32, pos, xf, xb);

  // ---- transformer layers ----
  const size_t DD = (size_t)CD * CD;
  for (int l = 0; l < CL; ++l) {
    gemm(xb, CD, 0, wQ + l * DD, 0, bq + l * CD, 0, nullptr, qb,  CD, 0,
         MTOK, CD, CD, EPI_STORE_BF16, 1, false);
    gemm(xb, CD, 0, wK + l * DD, 0, bk + l * CD, 0, nullptr, kb_, CD, 0,
         MTOK, CD, CD, EPI_STORE_BF16, 1, false);
    gemm(xb, CD, 0, wV + l * DD, 0, bv + l * CD, 0, nullptr, vb_, CD, 0,
         MTOK, CD, CD, EPI_STORE_BF16, 1, false);
    attention_kernel<<<(unsigned)(CB * CH), blk, 0, stream>>>(qb, kb_, vb_, ctxb);
    gemm(ctxb, CD, 0, wO + l * DD, 0, bo + l * CD, 0, tmp32, nullptr, CD, 0,
         MTOK, CD, CD, EPI_STORE_F32, 1, false);
    ln_residual_kernel<<<(unsigned)(MTOK / 8), blk, 0, stream>>>(
        xf, tmp32, ln1_g + l * CD, ln1_b + l * CD, xb, MTOK);
    gemm(xb, CD, 0, wW1 + (size_t)l * CD * CDFF, 0, b1 + l * CDFF, 0,
         nullptr, ffn_bf, CDFF, 0, MTOK, CD, CDFF, EPI_GELU | EPI_STORE_BF16, 1, false);
    gemm(ffn_bf, CDFF, 0, wW2 + (size_t)l * CDFF * CD, 0, b2f + l * CD, 0,
         tmp32, nullptr, CD, 0, MTOK, CDFF, CD, EPI_STORE_F32, 1, false);
    ln_residual_kernel<<<(unsigned)(MTOK / 8), blk, 0, stream>>>(
        xf, tmp32, ln2_g + l * CD, ln2_b + l * CD, xb, MTOK);
  }

  // ---- per-carrier MLP heads (batched over z = carrier) ----
  const int NC = CN * CHID;
  const size_t totH = (size_t)CB * NC;
  gemm(xb, (long long)CN * CD, CD, wM1, (long long)CD * CHID, mb1, CHID,
       h1f, nullptr, (long long)CN * CHID, CHID, CB, CD, CHID, EPI_STORE_F32, CN, true);
  bn_stats_kernel<<<(unsigned)((NC + 255) / 256), blk, 0, stream>>>(h1f, mean_, rstd_, NC, CB);
  bn_apply_gelu_kernel<<<(unsigned)((totH + 255) / 256), blk, 0, stream>>>(
      h1f, mean_, rstd_, bn1_g, bn1_b, h1b, NC, totH);

  gemm(h1b, (long long)CN * CHID, CHID, wM2, (long long)CHID * CHID, mb2, CHID,
       h2f, nullptr, (long long)CN * CHID, CHID, CB, CHID, CHID, EPI_STORE_F32, CN, true);
  bn_stats_kernel<<<(unsigned)((NC + 255) / 256), blk, 0, stream>>>(h2f, mean_, rstd_, NC, CB);
  bn_apply_gelu_kernel<<<(unsigned)((totH + 255) / 256), blk, 0, stream>>>(
      h2f, mean_, rstd_, bn2_g, bn2_b, h2b, NC, totH);

  gemm(h2b, (long long)CN * CHID, CHID, wM3, (long long)CHID * COUT, mb3, COUT,
       hout, nullptr, (long long)CN * COUT, COUT, CB, CHID, COUT, EPI_STORE_F32, CN, true);

  out_transpose_kernel<<<(unsigned)(((size_t)CB * COUT * CN + 255) / 256), blk, 0, stream>>>(
      hout, (float*)d_out);
}
